// MyGNN_67224828117676
// MI455X (gfx1250) — matrix operand order repaired and checked by reading, not compiled, not run on hardware
//
#include <hip/hip_runtime.h>

// ---------------------------------------------------------------------------
// Fused GNN forward for MI455X (gfx1250, wave32, WMMA + TDM).
//  - prep kernel: pre-transpose/convert all weights to f16 into d_ws (L2-hot)
//  - main kernel: one workgroup = 16 batch rows; activations resident in LDS
//    as f16; all GEMMs on v_wmma_f32_16x16x32_f16 (f32 accum); weight tiles
//    staged global->LDS by the Tensor Data Mover (tensor_load_to_lds),
//    fenced with s_wait_tensorcnt.
// ---------------------------------------------------------------------------

#define HDIM 128
#define NUM_NODES 13
#define NUM_LAYERS 8
#define NUM_JOINTS 17
#define D_MAX 33
#define OBS_DIM 348
#define ROWS 16
#define THREADS 256

typedef _Float16 h2  __attribute__((ext_vector_type(2)));
typedef _Float16 h8  __attribute__((ext_vector_type(8)));
typedef _Float16 h16 __attribute__((ext_vector_type(16)));
typedef float    f8  __attribute__((ext_vector_type(8)));
typedef unsigned int u32x4 __attribute__((ext_vector_type(4)));
typedef int      i32x4v __attribute__((ext_vector_type(4)));
typedef int      i32x8v __attribute__((ext_vector_type(8)));

// ---- f16 weight arena in d_ws (element offsets, halves) ----
#define W1T_BASE 0                  // [13][128][64]  (K padded 33->64)  106496
#define W2T_BASE 106496             // [13][128][128]                   212992
#define WMT_BASE 319488             // [8][128][256]                    262144
#define WS_HALVES 581632

// ---- LDS layout (bytes); total 225792 of the 320KB WGP LDS ----
#define OFF_X    0                  // 13*16*128 f16 = 53248
#define OFF_Y    53248              // ping-pong / phase-A scratch
#define OFF_AGG  106496
#define OFF_WT   159744             // up to 128*256 f16 = 65536
#define OFF_BIAS 225280             // 128 f32
#define LDS_BYTES 225792

// Feature gather table (padded with OBS_DIM -> reads the zeroed pad column)
__constant__ int c_feat[NUM_NODES][D_MAX] = {
 {0,1,2,3,4,22,23,24,25,26,27,45,46,47,48,49,50,51,52,53,54,175,176,177,178,179,180,270,271,272,273,274,275},
 {5,6,28,29,55,56,57,58,59,60,61,62,63,64,181,182,183,184,185,186,276,277,278,279,280,281,348,348,348,348,348,348,348},
 {7,30,65,66,67,68,69,70,71,72,73,74,187,188,189,190,191,192,253,254,255,282,283,284,285,286,287,348,348,348,348,348,348},
 {8,9,10,11,31,32,33,34,75,76,77,78,79,80,81,82,83,84,193,194,195,196,197,198,256,257,258,288,289,290,291,292,293},
 {11,34,85,86,87,88,89,90,91,92,93,94,199,200,201,202,203,204,259,294,295,296,297,298,299,348,348,348,348,348,348,348,348},
 {95,96,97,98,99,100,101,102,103,104,205,206,207,208,209,210,300,301,302,303,304,305,348,348,348,348,348,348,348,348,348,348,348},
 {12,13,14,15,35,36,37,38,105,106,107,108,109,110,111,112,113,114,211,212,213,214,215,216,260,261,262,306,307,308,309,310,311},
 {15,38,115,116,117,118,119,120,121,122,123,124,217,218,219,220,221,222,263,312,313,314,315,316,317,348,348,348,348,348,348,348,348},
 {125,126,127,128,129,130,131,132,133,134,223,224,225,226,227,228,318,319,320,321,322,323,348,348,348,348,348,348,348,348,348,348,348},
 {16,17,18,39,40,41,135,136,137,138,139,140,141,142,143,144,229,230,231,232,233,234,264,265,324,325,326,327,328,329,348,348,348},
 {18,41,145,146,147,148,149,150,151,152,153,154,235,236,237,238,239,240,266,330,331,332,333,334,335,348,348,348,348,348,348,348,348},
 {19,20,21,42,43,44,155,156,157,158,159,160,161,162,163,164,241,242,243,244,245,246,267,268,336,337,338,339,340,341,348,348,348},
 {21,44,165,166,167,168,169,170,171,172,173,174,247,248,249,250,251,252,269,342,343,344,345,346,347,348,348,348,348,348,348,348,348}
};
__constant__ int c_nbr_off[NUM_NODES + 1] = {0,3,5,8,10,12,13,15,17,18,20,21,23,24};
__constant__ int c_nbr[24] = {1,9,11, 0,2, 1,3,6, 2,4, 3,5, 4, 2,7, 6,8, 7, 0,10, 9, 0,12, 11};
__constant__ int c_ja[NUM_JOINTS] = {1,1,1,2,2,2,3,2,2,2,6,0,0,9,0,0,11};
__constant__ int c_jb[NUM_JOINTS] = {2,2,2,3,3,3,4,6,6,6,7,9,9,10,11,11,12};

// ---- TDM: 1-row f16 tile, global -> LDS (D# per ISA ch.8; type=2, 2B elems) ----
// 6-arg builtin form (this toolchain): (g0, g1, g2, g3, g_extra, cpol)
__device__ __forceinline__ void tdm_load_f16(const _Float16* gsrc, unsigned lds_off,
                                             unsigned nhalves) {
  unsigned long long ga = (unsigned long long)(const void*)gsrc;
  u32x4 g0;
  g0[0] = 1u;                                            // count=1, user descriptor
  g0[1] = lds_off;                                       // lds_addr (bytes)
  g0[2] = (unsigned)ga;                                  // global_addr[31:0]
  g0[3] = (unsigned)((ga >> 32) & 0x01FFFFFFull) | (2u << 30);  // ga[56:32] | type=2
  i32x8v g1;
  g1[0] = 1 << 16;                                       // data_size = 2 bytes
  g1[1] = (int)((nhalves & 0xFFFFu) << 16);              // tensor_dim0[15:0]
  g1[2] = (int)((nhalves >> 16) | (1u << 16));           // tensor_dim0[31:16] | tensor_dim1=1
  g1[3] = (int)((nhalves & 0xFFFFu) << 16);              // tile_dim0 = nhalves
  g1[4] = 0;                                             // tile_dim1/2 unused
  g1[5] = (int)nhalves;                                  // tensor_dim0_stride[31:0]
  g1[6] = 0;
  g1[7] = 0;
  i32x4v z4 = {0, 0, 0, 0};
  i32x8v z8 = {0, 0, 0, 0, 0, 0, 0, 0};
  __builtin_amdgcn_tensor_load_to_lds(g0, g1, z4, z4, z8, 0);
}
__device__ __forceinline__ void tdm_wait0() {
  __builtin_amdgcn_s_wait_tensorcnt((short)0);
}

// A fragment (16x32 f16): lane l holds row (l&15); halves 0..7 -> K=kb+khi+i,
// halves 8..15 -> K=kb+16+khi+i, khi=(l>>4)*8.   (ISA 7.12.2 A layout)
__device__ __forceinline__ h16 load_fragA(const _Float16* base, int m, int stride,
                                          int kb, int khi) {
  const h8* p0 = (const h8*)(base + m * stride + kb + khi);
  const h8* p1 = (const h8*)(base + m * stride + kb + 16 + khi);
  h8 a0 = *p0, a1 = *p1;
  h16 r;
#pragma unroll
  for (int i = 0; i < 8; ++i) { r[i] = a0[i]; r[i + 8] = a1[i]; }
  return r;
}
// B fragment (32x16 f16), weights N-major in LDS: lane l holds col (l&15);
// halves 0..7 -> K = kf + (l>>4)*8 + i, halves 8..15 -> K = kf + 16 + (l>>4)*8 + i
__device__ __forceinline__ h16 load_fragB(const _Float16* base, int ncol, int stride,
                                          int kf, int lane) {
  const h8* p = (const h8*)(base + ncol * stride + kf + ((lane >> 4) << 3));
  h8 b0 = p[0], b1v = p[2];
  h16 r;
#pragma unroll
  for (int i = 0; i < 8; ++i) { r[i] = b0[i]; r[i + 8] = b1v[i]; }
  return r;
}
__device__ __forceinline__ f8 wmma_f16(h16 a, h16 b, f8 c) {
  return __builtin_amdgcn_wmma_f32_16x16x32_f16(false, a, false, b, (short)0, c,
                                                false, false);
}
__device__ __forceinline__ float elu(float v) {
  return v > 0.f ? v : (__expf(v) - 1.f);
}

// ------------------- prep: weights -> transposed f16 arena -------------------
__global__ __launch_bounds__(THREADS)
void prep_weights_kernel(const float* __restrict__ W1, const float* __restrict__ W2,
                         const float* __restrict__ Wmsg, _Float16* __restrict__ ws) {
  int idx = blockIdx.x * THREADS + threadIdx.x;
  if (idx >= WS_HALVES) return;
  float v;
  if (idx < W2T_BASE) {                       // W1^T: [n][ho][d64], d padded 33->64
    int n = idx >> 13, r = idx & 8191;
    int ho = r >> 6, d = r & 63;
    v = (d < D_MAX) ? W1[(n * D_MAX + d) * HDIM + ho] : 0.f;
  } else if (idx < WMT_BASE) {                // W2^T: [n][ko][hi]
    int j = idx - W2T_BASE;
    int n = j >> 14, r = j & 16383;
    int ko = r >> 7, hi = r & 127;
    v = W2[(n * HDIM + hi) * HDIM + ko];
  } else {                                    // Wmsg^T: [l][no][k]
    int j = idx - WMT_BASE;
    int l = j >> 15, r = j & 32767;
    int no = r >> 8, k = r & 255;
    v = Wmsg[(l * 2 * HDIM + k) * HDIM + no];
  }
  ws[idx] = (_Float16)v;
}

// ------------------------------- main kernel --------------------------------
__global__ __launch_bounds__(THREADS, 1)
void gnn_fused_kernel(const float* __restrict__ obs,
                      const float* __restrict__ b1, const float* __restrict__ b2,
                      const float* __restrict__ bmsg,
                      const float* __restrict__ Wout, const float* __restrict__ bout,
                      const _Float16* __restrict__ wsw,
                      float* __restrict__ out, int batch) {
  extern __shared__ char smem[];
  _Float16* s_x    = (_Float16*)(smem + OFF_X);
  _Float16* s_y    = (_Float16*)(smem + OFF_Y);
  _Float16* s_agg  = (_Float16*)(smem + OFF_AGG);
  _Float16* s_wt   = (_Float16*)(smem + OFF_WT);
  float*    s_bias = (float*)(smem + OFF_BIAS);
  // phase-A scratch inside the ping-pong region (unused until layer phase)
  float*    s_obs  = (float*)(smem + OFF_Y);                          // 16*352 f32
  _Float16* s_h    = (_Float16*)(smem + OFF_Y + ROWS * 352 * 4);      // 16*128 f16
  _Float16* s_xn   = (_Float16*)(smem + OFF_Y + ROWS * 352 * 4 + ROWS * HDIM * 2);

  const int tid  = threadIdx.x;
  const int w    = tid >> 5;
  const int lane = tid & 31;
  const int l15  = lane & 15;
  const int khi  = (lane >> 4) * 8;   // A K-group offset; also D-matrix row base
  const int r0   = blockIdx.x * ROWS;

  // ---- stage obs rows, zero-padded to 352 cols (col>=348 reads as 0) ----
  for (int i = tid; i < ROWS * 352; i += THREADS) {
    int m = i / 352, c = i - m * 352;
    int row = r0 + m;
    float v = 0.f;
    if (c < OBS_DIM && row < batch) v = obs[row * OBS_DIM + c];
    s_obs[i] = v;
  }
  __syncthreads();

  // ================= Phase A: per-node encoders =================
  for (int n = 0; n < NUM_NODES; ++n) {
    // TDM: W1^T[n] (128x64 f16, 16KB) -> LDS, overlapped with the gather
    if (w == 0) tdm_load_f16(wsw + W1T_BASE + n * 8192, OFF_WT, 8192);
    for (int i = tid; i < ROWS * 64; i += THREADS) {
      int m = i >> 6, d = i & 63;
      int fi = (d < D_MAX) ? c_feat[n][d] : OBS_DIM;
      s_xn[m * 64 + d] = (_Float16)s_obs[m * 352 + fi];
    }
    if (w == 0) tdm_wait0();
    __syncthreads();
    // h = elu(xn @ W1 + b1); each wave owns one 16-wide N tile
    {
      const int nt = w;
      f8 acc = {};
#pragma unroll
      for (int ks = 0; ks < 2; ++ks) {
        h16 a  = load_fragA(s_xn, l15, 64, ks * 32, khi);
        h16 bf = load_fragB(s_wt, nt * 16 + l15, 64, ks * 32, lane);
        acc = wmma_f16(a, bf, acc);
      }
      const int col = nt * 16 + l15;
      const float bias = b1[n * HDIM + col];
#pragma unroll
      for (int r = 0; r < 8; ++r)
        s_h[(khi + r) * HDIM + col] = (_Float16)elu(acc[r] + bias);
    }
    __syncthreads();
    // TDM: W2^T[n] (128x128 f16, 32KB) -> LDS
    if (w == 0) { tdm_load_f16(wsw + W2T_BASE + n * 16384, OFF_WT, 16384); tdm_wait0(); }
    __syncthreads();
    // x0[n] = h @ W2 + b2  (no activation)
    {
      const int nt = w;
      f8 acc = {};
#pragma unroll
      for (int ks = 0; ks < 4; ++ks) {
        h16 a  = load_fragA(s_h, l15, HDIM, ks * 32, khi);
        h16 bf = load_fragB(s_wt, nt * 16 + l15, HDIM, ks * 32, lane);
        acc = wmma_f16(a, bf, acc);
      }
      const int col = nt * 16 + l15;
      const float bias = b2[n * HDIM + col];
#pragma unroll
      for (int r = 0; r < 8; ++r)
        s_x[n * (ROWS * HDIM) + (khi + r) * HDIM + col] = (_Float16)(acc[r] + bias);
    }
    __syncthreads();
  }

  // ================= Phase B: 8 message-passing layers =================
  for (int l = 0; l < NUM_LAYERS; ++l) {
    // TDM: Wmsg^T[l] (128x256 f16, 64KB) -> LDS; overlap with bias + agg
    if (w == 0) tdm_load_f16(wsw + WMT_BASE + l * 32768, OFF_WT, 32768);
    if (tid < HDIM) s_bias[tid] = bmsg[l * HDIM + tid];
    if (l + 1 < NUM_LAYERS)  // warm next layer's f16 weights (global_prefetch_b8)
      __builtin_prefetch(wsw + WMT_BASE + (l + 1) * 32768 + tid * HDIM, 0, 1);
    // agg[d] = sum_{s in nbr(d)} x[s]  -- packed f16 adds (v_pk_add_f16)
    {
      const h2* xs = (const h2*)s_x;
      h2* ag = (h2*)s_agg;
      for (int i = tid; i < NUM_NODES * ROWS * HDIM / 2; i += THREADS) {
        int d = i >> 10, rem = i & 1023;
        int e = c_nbr_off[d];
        h2 s = xs[c_nbr[e] * 1024 + rem];
        for (++e; e < c_nbr_off[d + 1]; ++e) s += xs[c_nbr[e] * 1024 + rem];
        ag[i] = s;
      }
    }
    if (w == 0) tdm_wait0();
    __syncthreads();
    // x_new = elu([x | agg] @ Wmsg + b); wave w owns N-tile w for all 13 nodes,
    // so its 8 B fragments are loaded once and reused 13x.
    {
      const int nt = w;
      h16 bfr[8];
#pragma unroll
      for (int ks = 0; ks < 8; ++ks)
        bfr[ks] = load_fragB(s_wt, nt * 16 + l15, 2 * HDIM, ks * 32, lane);
      const int col = nt * 16 + l15;
      const float bias = s_bias[col];
      for (int n = 0; n < NUM_NODES; ++n) {
        f8 acc = {};
#pragma unroll
        for (int ks = 0; ks < 8; ++ks) {
          const int kf = ks * 32;
          const _Float16* asrc = (kf < HDIM) ? (s_x + n * 2048) : (s_agg + n * 2048);
          h16 a = load_fragA(asrc, l15, HDIM, kf & (HDIM - 1), khi);
          acc = wmma_f16(a, bfr[ks], acc);
        }
#pragma unroll
        for (int r = 0; r < 8; ++r)
          s_y[n * 2048 + (khi + r) * HDIM + col] = (_Float16)elu(acc[r] + bias);
      }
    }
    __syncthreads();
    _Float16* tp = s_x; s_x = s_y; s_y = tp;
  }

  // ================= Output: per-joint dot products =================
  for (int i = tid; i < ROWS * NUM_JOINTS; i += THREADS) {
    int m = i / NUM_JOINTS, j = i - m * NUM_JOINTS;
    int row = r0 + m;
    if (row >= batch) continue;
    const _Float16* xa = s_x + c_ja[j] * 2048 + m * HDIM;
    const _Float16* xb = s_x + c_jb[j] * 2048 + m * HDIM;
    const float* wo = Wout + j * 2 * HDIM;
    float acc = bout[j];
#pragma unroll 4
    for (int f = 0; f < HDIM; ++f)
      acc += (float)xa[f] * wo[f] + (float)xb[f] * wo[HDIM + f];
    out[row * NUM_JOINTS + j] = acc;
  }
}

extern "C" void kernel_launch(void* const* d_in, const int* in_sizes, int n_in,
                              void* d_out, int out_size, void* d_ws, size_t ws_size,
                              hipStream_t stream) {
  (void)n_in; (void)out_size; (void)ws_size;
  const float* obs  = (const float*)d_in[0];
  const float* W1   = (const float*)d_in[1];
  const float* b1   = (const float*)d_in[2];
  const float* W2   = (const float*)d_in[3];
  const float* b2   = (const float*)d_in[4];
  const float* Wmsg = (const float*)d_in[5];
  const float* bmsg = (const float*)d_in[6];
  const float* Wout = (const float*)d_in[7];
  const float* bout = (const float*)d_in[8];
  float* out = (float*)d_out;
  _Float16* ws = (_Float16*)d_ws;

  const int prep_blocks = (WS_HALVES + THREADS - 1) / THREADS;
  hipLaunchKernelGGL(prep_weights_kernel, dim3(prep_blocks), dim3(THREADS), 0, stream,
                     W1, W2, Wmsg, ws);

  const int batch  = in_sizes[0] / OBS_DIM;
  const int blocks = (batch + ROWS - 1) / ROWS;
  hipLaunchKernelGGL(gnn_fused_kernel, dim3(blocks), dim3(THREADS), LDS_BYTES, stream,
                     obs, b1, b2, bmsg, Wout, bout, ws, out, batch);
}
